// Attention_10642928960150
// MI455X (gfx1250) — matrix operand order repaired and checked by reading, not compiled, not run on hardware
//
#include <hip/hip_runtime.h>
#include <stdint.h>

// ---------------------------------------------------------------------------
// Problem constants (from reference)
// ---------------------------------------------------------------------------
#define DIM      4096
#define N_HEADS  32
#define HEAD     128          // DIM / N_HEADS
#define SEQ      2048
#define BSZ      2
#define MTOT     4096         // BSZ * SEQ
#define NELT     16777216ull  // 4096 * 4096 elements (one activation / weight)

typedef __bf16 bf16_t;
typedef __bf16 v16bf __attribute__((ext_vector_type(16)));
typedef float  v8f   __attribute__((ext_vector_type(8)));
typedef unsigned int u32x4 __attribute__((ext_vector_type(4)));
typedef int          i32x4 __attribute__((ext_vector_type(4)));
typedef int          i32x8 __attribute__((ext_vector_type(8)));

union Frag16 {            // 16 bf16 = 8 VGPRs = one WMMA A/B fragment
    bf16_t h[16];
    uint4  q[2];
};

static __device__ inline v8f wmma_bf16(v16bf a, v16bf b, v8f c) {
    // D = A(16x32 bf16) * B(32x16 bf16) + C(16x16 f32)
    return __builtin_amdgcn_wmma_f32_16x16x32_bf16(
        /*neg_a=*/false, a, /*neg_b=*/false, b,
        /*c_mod=*/(short)0, c, /*reuse_a=*/false, /*reuse_b=*/false);
}

// ---------------------------------------------------------------------------
// TDM: issue a 2-D tensor_load_to_lds of a 32(k) x 128(rows) fp32 tile.
// D# per CDNA5 ISA 8.3-8.5. Padding: +4 DWORDs per 32-DWORD row -> LDS row
// stride 36 words (144B: 16B-aligned rows, bank-skewed half-wave reads).
// Toolchain here is the 6-arg builtin: (g0, g1, g2, g3, g4, cpol).
// ---------------------------------------------------------------------------
#define LDSW 36   // fp32 words per LDS tile row after TDM padding

static __device__ inline void tdm_load_tile(const float* gtile, unsigned lds_off) {
    const unsigned long long ga = (unsigned long long)(uintptr_t)gtile;
    u32x4 g0;
    g0[0] = 1u;                                   // count=1, user descriptor
    g0[1] = lds_off;                              // lds_addr [63:32]
    g0[2] = (unsigned)ga;                         // global_addr lo
    g0[3] = (unsigned)((ga >> 32) & 0x01FFFFFFull) | 0x80000000u; // addr hi | type=2

    i32x8 g1;
    g1[0] = (2 << 16)        // data_size = 4 bytes
          | (1 << 20)        // pad_enable
          | (4 << 22)        // pad_interval: 2^(4+1)=32 DWORDs (one tile row)
          | (3 << 25);       // pad_amount: 3+1 = 4 DWORDs
    g1[1] = (int)(4096u << 16);   // tensor_dim0[15:0]  at [63:48]  (=4096)
    g1[2] = (int)(4096u << 16);   // dim0 hi16=0; tensor_dim1[15:0] at [95:80]
    g1[3] = (int)(32u << 16);     // dim1 hi16=0; tile_dim0=32 at [127:112]
    g1[4] = 128;                  // tile_dim1=128 at [143:128]; tile_dim2=0
    g1[5] = 4096;                 // tensor_dim0_stride[31:0] = 4096 elements
    g1[6] = 0;                    // stride hi / tensor_dim1_stride lo
    g1[7] = 0;
    const i32x4 z4 = {0, 0, 0, 0};                // groups 2/3: NULL (<=2D tensor)
    const i32x8 z8 = {0, 0, 0, 0, 0, 0, 0, 0};    // trailing group: unused
    __builtin_amdgcn_tensor_load_to_lds(g0, g1, z4, z4, z8, 0);
}

// Build one bf16 WMMA fragment from two 8-float LDS chunks (fp32 -> bf16).
static __device__ inline v16bf frag_from_f32(const float* p0, const float* p1) {
    const float4 x0 = ((const float4*)p0)[0];
    const float4 x1 = ((const float4*)p0)[1];
    const float4 y0 = ((const float4*)p1)[0];
    const float4 y1 = ((const float4*)p1)[1];
    Frag16 f;
    f.h[0]=(bf16_t)x0.x; f.h[1]=(bf16_t)x0.y; f.h[2]=(bf16_t)x0.z; f.h[3]=(bf16_t)x0.w;
    f.h[4]=(bf16_t)x1.x; f.h[5]=(bf16_t)x1.y; f.h[6]=(bf16_t)x1.z; f.h[7]=(bf16_t)x1.w;
    f.h[8]=(bf16_t)y0.x; f.h[9]=(bf16_t)y0.y; f.h[10]=(bf16_t)y0.z; f.h[11]=(bf16_t)y0.w;
    f.h[12]=(bf16_t)y1.x; f.h[13]=(bf16_t)y1.y; f.h[14]=(bf16_t)y1.z; f.h[15]=(bf16_t)y1.w;
    return *(const v16bf*)&f;
}

// ---------------------------------------------------------------------------
// GEMM (NT): C[m,n] = sum_k A[m,k] * B[n,k]; fp32 operands, bf16 WMMA math.
// 128x128 workgroup tile, 8 waves of 32x64, K-step 32.
// TDM double-buffered staging: wave 0 issues next tiles, s_wait_tensorcnt 2,
// workgroup barrier gates compute. fp32->bf16 at fragment-build time.
// ---------------------------------------------------------------------------
__global__ __launch_bounds__(256) void gemm_nt_tdm(
        const float* __restrict__ A, const float* __restrict__ B,
        float* __restrict__ C) {
    __shared__ __align__(16) float ls[2][2][128 * LDSW];   // [buf][A/B] = 72KB

    const int tid  = threadIdx.x;
    const int lane = tid & 31;
    const int wv   = tid >> 5;            // 8 waves
    const int half = lane >> 4;           // lane-half selects K-chunk per ISA layout
    const int l16  = lane & 15;

    const int tm = blockIdx.y * 128;
    const int tn = blockIdx.x * 128;
    const int wm = (wv & 3) * 32;         // 4 waves along M
    const int wn = (wv >> 2) * 64;        // 2 waves along N

    v8f acc[2][4];
    const v8f vzero = {0.f,0.f,0.f,0.f,0.f,0.f,0.f,0.f};
#pragma unroll
    for (int i = 0; i < 2; ++i)
#pragma unroll
        for (int j = 0; j < 4; ++j) acc[i][j] = vzero;

    const float* Abase = A + (size_t)tm * DIM;   // tile row 0, k advances by 32
    const float* Bbase = B + (size_t)tn * DIM;

    const int nsteps = DIM / 32;
    if (wv == 0) {                                // prologue: stage step 0
        tdm_load_tile(Abase, (unsigned)(uintptr_t)&ls[0][0][0]);
        tdm_load_tile(Bbase, (unsigned)(uintptr_t)&ls[0][1][0]);
    }

    for (int i = 0; i < nsteps; ++i) {
        const int cur = i & 1;
        if (wv == 0) {
            if (i + 1 < nsteps) {                 // stage step i+1 into other buffer
                const int nxt = cur ^ 1;
                tdm_load_tile(Abase + (i + 1) * 32, (unsigned)(uintptr_t)&ls[nxt][0][0]);
                tdm_load_tile(Bbase + (i + 1) * 32, (unsigned)(uintptr_t)&ls[nxt][1][0]);
                __builtin_amdgcn_s_wait_tensorcnt(2);  // in-order: step i tiles done
            } else {
                __builtin_amdgcn_s_wait_tensorcnt(0);
            }
        }
        __syncthreads();

        const float* Ab = &ls[cur][0][0];
        const float* Bb = &ls[cur][1][0];

        // A fragments: lanes 0-15 hold K 0-7 & 16-23, lanes 16-31 hold K 8-15 & 24-31
        v16bf af[2];
#pragma unroll
        for (int sm = 0; sm < 2; ++sm) {
            const int row = wm + sm * 16 + l16;
            const int c0  = half * 8;
            af[sm] = frag_from_f32(Ab + row * LDSW + c0, Ab + row * LDSW + c0 + 16);
        }
        // B fragments: lanes 0-15 hold K 0-7 & 16-23, lanes 16-31 hold K 8-15 & 24-31
        v16bf bfq[4];
#pragma unroll
        for (int sn = 0; sn < 4; ++sn) {
            const int row = wn + sn * 16 + l16;
            const int c0  = half * 8;
            bfq[sn] = frag_from_f32(Bb + row * LDSW + c0, Bb + row * LDSW + c0 + 16);
        }
#pragma unroll
        for (int sm = 0; sm < 2; ++sm)
#pragma unroll
            for (int sn = 0; sn < 4; ++sn)
                acc[sm][sn] = wmma_bf16(af[sm], bfq[sn], acc[sm][sn]);
        __syncthreads();   // all waves done reading before buffer is re-staged
    }

    // C layout: VGPR r -> row r (lanes 0-15) / r+8 (lanes 16-31), col = lane%16
#pragma unroll
    for (int sm = 0; sm < 2; ++sm)
#pragma unroll
        for (int sn = 0; sn < 4; ++sn)
#pragma unroll
            for (int r = 0; r < 8; ++r) {
                const int row = tm + wm + sm * 16 + r + 8 * half;
                const int col = tn + wn + sn * 16 + l16;
                C[(size_t)row * DIM + col] = acc[sm][sn][r];
            }
}

// ---------------------------------------------------------------------------
// RoPE. Pair index i enumerates (b, s, h, d/2) in order -> flat offset 2*i.
// ---------------------------------------------------------------------------
#define NPAIRS (BSZ * SEQ * N_HEADS * (HEAD / 2))

__global__ void rope_q_kernel(const float* __restrict__ q,
                              const float* __restrict__ cosT,
                              const float* __restrict__ sinT,
                              bf16_t* __restrict__ qb) {
    size_t i = (size_t)blockIdx.x * blockDim.x + threadIdx.x;
    if (i >= (size_t)NPAIRS) return;
    const int pi = (int)(i & 63);
    const int s  = (int)((i >> (6 + 5)) & (SEQ - 1));   // skip head bits (5)
    const float c  = cosT[s * 64 + pi];
    const float sn = sinT[s * 64 + pi];
    const size_t base = i * 2;
    const float a = q[base], b = q[base + 1];
    const float scale = 0.08838834764831845f;           // 1/sqrt(HEAD), folded into Q
    qb[base]     = (bf16_t)((a * c - b * sn) * scale);
    qb[base + 1] = (bf16_t)((a * sn + b * c) * scale);
}

__global__ void rope_k_kernel(float* __restrict__ k,
                              const float* __restrict__ cosT,
                              const float* __restrict__ sinT,
                              bf16_t* __restrict__ kb) {
    size_t i = (size_t)blockIdx.x * blockDim.x + threadIdx.x;
    if (i >= (size_t)NPAIRS) return;
    const int pi = (int)(i & 63);
    const int s  = (int)((i >> (6 + 5)) & (SEQ - 1));
    const float c  = cosT[s * 64 + pi];
    const float sn = sinT[s * 64 + pi];
    const size_t base = i * 2;
    const float a = k[base], b = k[base + 1];
    const float r0 = a * c - b * sn;
    const float r1 = a * sn + b * c;
    k[base] = r0; k[base + 1] = r1;                     // fp32 `keys` output
    kb[base] = (bf16_t)r0; kb[base + 1] = (bf16_t)r1;   // bf16 copy for attention
}

// ---------------------------------------------------------------------------
// V transpose: vt[b][h][d][s] = (bf16) v[b][s][h][d]  (per-head 128 x 2048)
// so P*V B-fragments read contiguous 8-element runs along the K(=seq) dim.
// ---------------------------------------------------------------------------
__global__ __launch_bounds__(256) void transpose_v_kernel(
        const float* __restrict__ v, bf16_t* __restrict__ vt) {
    __shared__ float tile[32][33];
    const int b  = blockIdx.z >> 5;
    const int h  = blockIdx.z & 31;
    const int s0 = blockIdx.x * 32;
    const int d0 = blockIdx.y * 32;
    const int tx = threadIdx.x & 31;
    const int ty = threadIdx.x >> 5;      // 8 rows per step
#pragma unroll
    for (int r = ty; r < 32; r += 8)
        tile[r][tx] = v[(size_t)(b * SEQ + s0 + r) * DIM + h * HEAD + d0 + tx];
    __syncthreads();
#pragma unroll
    for (int r = ty; r < 32; r += 8)
        vt[(((size_t)(b * N_HEADS + h) * HEAD) + d0 + r) * SEQ + s0 + tx] =
            (bf16_t)tile[tx][r];
}

// ---------------------------------------------------------------------------
// Flash attention: one wave per (b, h, 16-row Q tile); KV tiles of 32.
// S via 8 WMMAs (4 K-steps x 2 column halves), online softmax with half-wave
// shfl reductions, P C-layout -> A-layout through a 1KB LDS tile, O via 8 WMMAs.
// ---------------------------------------------------------------------------
#define NEG_INF (-3.0e38f)
#define PSTRIDE 40   // bf16 row stride for the P tile (16B-aligned, bank-skewed)

__global__ __launch_bounds__(32) void flash_attn_kernel(
        const bf16_t* __restrict__ Qb,   // (b,s,h,d) bf16, pre-scaled by 1/sqrt(d)
        const bf16_t* __restrict__ Kb,   // (b,s,h,d) bf16, RoPE'd
        const bf16_t* __restrict__ Vt,   // (b,h,d,s) bf16
        const int*    __restrict__ causal_flag,
        float* __restrict__ attn) {      // (b,s,h,d) fp32
    __shared__ __align__(16) bf16_t lp[16 * PSTRIDE];

    const int lane = threadIdx.x;
    const int l16  = lane & 15;
    const int half = lane >> 4;
    const int q0   = blockIdx.x * 16;
    const int h    = blockIdx.y;
    const int b    = blockIdx.z;
    const int causal = causal_flag[0];

    // Q fragments for the 4 K-steps over head_dim=128
    Frag16 qf[4];
    const bf16_t* qbase = Qb + ((size_t)(b * SEQ + q0 + l16) * DIM + h * HEAD);
#pragma unroll
    for (int kk = 0; kk < 4; ++kk) {
        const int c0 = kk * 32 + half * 8;
        qf[kk].q[0] = *(const uint4*)(qbase + c0);
        qf[kk].q[1] = *(const uint4*)(qbase + c0 + 16);
    }

    const v8f vzero = {0.f,0.f,0.f,0.f,0.f,0.f,0.f,0.f};
    v8f o[8];
#pragma unroll
    for (int d = 0; d < 8; ++d) o[d] = vzero;
    float mrow[8], lrow[8];
#pragma unroll
    for (int r = 0; r < 8; ++r) { mrow[r] = NEG_INF; lrow[r] = 0.f; }

    const int jlast = causal ? (q0 + 15) : (SEQ - 1);
    for (int j0 = 0; j0 <= jlast; j0 += 32) {
        // ---- S tile 16x32: columns [j0, j0+32) -----------------------------
        v8f s0 = vzero, s1 = vzero;
#pragma unroll
        for (int kk = 0; kk < 4; ++kk) {
            const int d = kk * 32 + half * 8;     // 8 bf16 at d and 8 at d+16 per lane
            const bf16_t* k0p = Kb + ((size_t)(b * SEQ + j0 +      l16) * DIM + h * HEAD + d);
            const bf16_t* k1p = Kb + ((size_t)(b * SEQ + j0 + 16 + l16) * DIM + h * HEAD + d);
            Frag16 kb0, kb1;
            kb0.q[0] = *(const uint4*)k0p; kb0.q[1] = *(const uint4*)(k0p + 16);
            kb1.q[0] = *(const uint4*)k1p; kb1.q[1] = *(const uint4*)(k1p + 16);
            s0 = wmma_bf16(*(const v16bf*)&qf[kk], *(const v16bf*)&kb0, s0);
            s1 = wmma_bf16(*(const v16bf*)&qf[kk], *(const v16bf*)&kb1, s1);
        }

        // ---- causal mask on diagonal-straddling tiles ----------------------
        if (causal && (j0 + 31 > q0)) {
#pragma unroll
            for (int r = 0; r < 8; ++r) {
                const int qg = q0 + r + 8 * half;
                if (j0 +      l16 > qg) s0[r] = NEG_INF;
                if (j0 + 16 + l16 > qg) s1[r] = NEG_INF;
            }
        }

        // ---- online softmax (row = (r, lane-half); columns across 16 lanes)
#pragma unroll
        for (int r = 0; r < 8; ++r) {
            float mx = fmaxf(s0[r], s1[r]);
#pragma unroll
            for (int m = 1; m < 16; m <<= 1) mx = fmaxf(mx, __shfl_xor(mx, m, 32));
            const float mn = fmaxf(mrow[r], mx);
            const float alpha = __expf(mrow[r] - mn);
            mrow[r] = mn;
            const float p0 = __expf(s0[r] - mn);
            const float p1 = __expf(s1[r] - mn);
            float ps = p0 + p1;
#pragma unroll
            for (int m = 1; m < 16; m <<= 1) ps += __shfl_xor(ps, m, 32);
            lrow[r] = lrow[r] * alpha + ps;
            lp[(r + 8 * half) * PSTRIDE +      l16] = (bf16_t)p0;
            lp[(r + 8 * half) * PSTRIDE + 16 + l16] = (bf16_t)p1;
#pragma unroll
            for (int d = 0; d < 8; ++d) o[d][r] *= alpha;
        }
        __syncthreads();

        // ---- P (A layout) and O += P * V -----------------------------------
        Frag16 pf;
        {
            const int c0 = half * 8;
            pf.q[0] = *(const uint4*)&lp[l16 * PSTRIDE + c0];
            pf.q[1] = *(const uint4*)&lp[l16 * PSTRIDE + c0 + 16];
        }
#pragma unroll
        for (int di = 0; di < 8; ++di) {
            const int d = di * 16 + l16;
            const bf16_t* vp = Vt + (((size_t)(b * N_HEADS + h) * HEAD + d) * SEQ
                                     + j0 + half * 8);
            Frag16 vf;
            vf.q[0] = *(const uint4*)vp; vf.q[1] = *(const uint4*)(vp + 16);
            o[di] = wmma_bf16(*(const v16bf*)&pf, *(const v16bf*)&vf, o[di]);
        }
        __syncthreads();
    }

    // ---- epilogue: O / l, write (b,s,h,d) ---------------------------------
#pragma unroll
    for (int r = 0; r < 8; ++r) {
        const float inv = 1.0f / lrow[r];
        const int qg = q0 + r + 8 * half;
#pragma unroll
        for (int di = 0; di < 8; ++di)
            attn[(size_t)(b * SEQ + qg) * DIM + h * HEAD + di * 16 + l16] =
                o[di][r] * inv;
    }
}

// ---------------------------------------------------------------------------
// Host orchestration
// ---------------------------------------------------------------------------
extern "C" void kernel_launch(void* const* d_in, const int* in_sizes, int n_in,
                              void* d_out, int out_size, void* d_ws, size_t ws_size,
                              hipStream_t stream) {
    const float* x    = (const float*)d_in[0];
    const float* fcos = (const float*)d_in[1];
    const float* fsin = (const float*)d_in[2];
    const float* wq   = (const float*)d_in[3];
    const float* wk   = (const float*)d_in[4];
    const float* wv   = (const float*)d_in[5];
    const float* wo   = (const float*)d_in[6];
    const int*   cz   = (const int*)  d_in[7];

    float* out    = (float*)d_out;          // (b,s,DIM)
    float* keys   = out  + NELT;            // (b,s,h,d) == contiguous
    float* values = keys + NELT;

    char* ws = (char*)d_ws;
    float*  qf32 = (float*)ws;                        // 64 MiB; reused for attn
    bf16_t* qb   = (bf16_t*)(ws + (64ull  << 20));    // 32 MiB
    bf16_t* kb   = (bf16_t*)(ws + (96ull  << 20));    // 32 MiB
    bf16_t* vt   = (bf16_t*)(ws + (128ull << 20));    // 32 MiB

    const dim3 gg(DIM / 128, MTOT / 128);             // 32 x 32 workgroups

    // Projections (NT GEMMs): K and V land directly in the output buffer.
    gemm_nt_tdm<<<gg, 256, 0, stream>>>(x, wq, qf32);
    gemm_nt_tdm<<<gg, 256, 0, stream>>>(x, wk, keys);
    gemm_nt_tdm<<<gg, 256, 0, stream>>>(x, wv, values);

    const int rblocks = (NPAIRS + 255) / 256;
    rope_q_kernel<<<rblocks, 256, 0, stream>>>(qf32, fcos, fsin, qb);
    rope_k_kernel<<<rblocks, 256, 0, stream>>>(keys, fcos, fsin, kb);

    transpose_v_kernel<<<dim3(SEQ / 32, HEAD / 32, BSZ * N_HEADS), 256, 0, stream>>>(
        values, vt);

    // attn overwrites qf32 (Q fp32 is dead after rope_q).
    flash_attn_kernel<<<dim3(SEQ / 16, N_HEADS, BSZ), 32, 0, stream>>>(
        qb, kb, vt, cz, qf32);

    gemm_nt_tdm<<<gg, 256, 0, stream>>>(qf32, wo, out);
}
